// ConvCRF2d_26233660244508
// MI455X (gfx1250) — compile-verified, matrix operand order checked by reading, not run on hardware
//
#include <hip/hip_runtime.h>
#include <hip/hip_bf16.h>

#define IMG_H 256
#define IMG_W 256
#define NCLS  21
#define SPANR 3
#define KS    7
#define NTAP  49
#define NB    4
#define EPSV  1e-20f
#define MOM   0.8f
#define TS    16
#define THALO (TS + 2*SPANR)   // 22

typedef __attribute__((ext_vector_type(16))) _Float16 v16h;
typedef __attribute__((ext_vector_type(8)))  float    v8f;

// ---------------------------------------------------------------------------
// Kernel 1: data-dependent color Gaussian  g_col[b][n][r][c] (*col_compat)
// and its window-sum norm  normc[b][r][c] = rsqrt(sum_n g_raw + eps).
// Pairwise feature dot products computed with v_wmma_f32_16x16x32_f16:
//   A[p,k<3]  = f_k(r, c0+p)          (16 row pixels)
//   B[k<3,q]  = f_k(r+dr, c0-3+q)     (22 neighbor cols -> two 16-wide tiles)
//   D[p,q]    = f(p).f(q);  dist2 = S(p)+S(q)-2 D[p,q]
// One wave (32 lanes) per (b, row, 16-col segment).
// ---------------------------------------------------------------------------
__global__ __launch_bounds__(32)
void build_col_gaussian(const float* __restrict__ img,
                        const float* __restrict__ col_schan,
                        const float* __restrict__ col_compat,
                        float* __restrict__ gcol,
                        float* __restrict__ normc)
{
    __shared__ float sDt[16][24];   // D[p][q], q in [0,22)
    __shared__ float sSnb[24];      // |f(q)|^2
    __shared__ float sM[24];        // column-valid mask
    __shared__ float sF[3][24];     // neighbor features (f16-rounded, as f32)

    const int lane = threadIdx.x;
    const int c0   = blockIdx.x * 16;
    const int r    = blockIdx.y;
    const int b    = blockIdx.z;

    const float schan  = col_schan[0];
    const float compat = col_compat[0];
    const float* imgb  = img + (size_t)b * 3 * IMG_H * IMG_W;

    // center pixel features (lanes 0..15 own pixel p = lane)
    float Sc = 0.f;
    v16h A = {};
    if (lane < 16) {
        int cc = c0 + lane;
        float f0 = imgb[0*IMG_H*IMG_W + r*IMG_W + cc] * schan;
        float f1 = imgb[1*IMG_H*IMG_W + r*IMG_W + cc] * schan;
        float f2 = imgb[2*IMG_H*IMG_W + r*IMG_W + cc] * schan;
        _Float16 h0 = (_Float16)f0, h1 = (_Float16)f1, h2 = (_Float16)f2;
        float g0 = (float)h0, g1 = (float)h1, g2 = (float)h2;
        Sc = g0*g0 + g1*g1 + g2*g2;
        A[0] = h0; A[1] = h1; A[2] = h2;   // K=0..2; lanes>=16 carry K>=8 -> all zero
    }

    float normsum = 0.f;

    for (int dr = -SPANR; dr <= SPANR; ++dr) {
        int rr = r + dr;                       // uniform across wave
        if (rr < 0 || rr >= IMG_H) {
            if (lane < 16) {
                for (int dc = -SPANR; dc <= SPANR; ++dc) {
                    int n = (dr+SPANR)*KS + (dc+SPANR);
                    gcol[(((size_t)b*NTAP + n)*IMG_H + r)*IMG_W + c0 + lane] = 0.f;
                }
            }
            continue;
        }
        // neighbor row features, lanes 0..21 -> q = lane, col = c0-3+q
        if (lane < 22) {
            int nc = c0 - SPANR + lane;
            float m = (nc >= 0 && nc < IMG_W) ? 1.f : 0.f;
            float e0 = 0.f, e1 = 0.f, e2 = 0.f;
            if (m != 0.f) {
                e0 = imgb[0*IMG_H*IMG_W + rr*IMG_W + nc] * schan;
                e1 = imgb[1*IMG_H*IMG_W + rr*IMG_W + nc] * schan;
                e2 = imgb[2*IMG_H*IMG_W + rr*IMG_W + nc] * schan;
            }
            _Float16 q0 = (_Float16)e0, q1 = (_Float16)e1, q2 = (_Float16)e2;
            float w0 = (float)q0, w1 = (float)q1, w2 = (float)q2;
            sF[0][lane] = w0; sF[1][lane] = w1; sF[2][lane] = w2;
            sSnb[lane]  = w0*w0 + w1*w1 + w2*w2;
            sM[lane]    = m;
        }
        __syncthreads();

        // B tiles: columns q = lane (tile1), q = lane+16 (tile2, only 6 valid)
        v16h B1 = {}, B2 = {};
        if (lane < 16) {
            B1[0] = (_Float16)sF[0][lane];
            B1[1] = (_Float16)sF[1][lane];
            B1[2] = (_Float16)sF[2][lane];
            if (lane < 6) {
                B2[0] = (_Float16)sF[0][lane+16];
                B2[1] = (_Float16)sF[1][lane+16];
                B2[2] = (_Float16)sF[2][lane+16];
            }
        }
        v8f d1 = {}; v8f d2 = {};
        d1 = __builtin_amdgcn_wmma_f32_16x16x32_f16(false, A, false, B1, (short)0, d1, false, false);
        d2 = __builtin_amdgcn_wmma_f32_16x16x32_f16(false, A, false, B2, (short)0, d2, false, false);

        // scatter D into LDS: lane holds N = lane&15, rows M = v + (lane<16 ? 0 : 8)
        {
            int N = lane & 15;
            int Mb = (lane < 16) ? 0 : 8;
            #pragma unroll
            for (int v = 0; v < 8; ++v) sDt[Mb + v][N] = d1[v];
            if (N < 6) {
                #pragma unroll
                for (int v = 0; v < 8; ++v) sDt[Mb + v][16 + N] = d2[v];
            }
        }
        __syncthreads();

        if (lane < 16) {
            #pragma unroll
            for (int dc = -SPANR; dc <= SPANR; ++dc) {
                int q = lane + SPANR + dc;
                float dist = Sc + sSnb[q] - 2.f * sDt[lane][q];
                dist = fmaxf(dist, 0.f);
                float graw = sM[q] * __expf(-0.5f * dist);
                normsum += graw;
                int n = (dr+SPANR)*KS + (dc+SPANR);
                gcol[(((size_t)b*NTAP + n)*IMG_H + r)*IMG_W + c0 + lane] = compat * graw;
            }
        }
        __syncthreads();
    }

    if (lane < 16) {
        normc[((size_t)b*IMG_H + r)*IMG_W + c0 + lane] = rsqrtf(normsum + EPSV);
    }
}

// ---------------------------------------------------------------------------
// Kernel 2: positional norm. g_pos is spatially constant (mesh is meshgrid),
// so the window sum is separable: normp = rsqrt(Sr(x)*Sc(y) + eps).
// ---------------------------------------------------------------------------
__global__ void build_pos_norm(const float* __restrict__ col_sdims,
                               float* __restrict__ normp)
{
    int idx = blockIdx.x * blockDim.x + threadIdx.x;
    if (idx >= IMG_H * IMG_W) return;
    int r = idx / IMG_W, c = idx % IMG_W;
    float s = col_sdims[0];
    float a = 0.5f * s * s;
    float ex[4] = {1.f, __expf(-a), __expf(-4.f*a), __expf(-9.f*a)};
    float Sr = 0.f, Scl = 0.f;
    #pragma unroll
    for (int d = -SPANR; d <= SPANR; ++d) {
        int ad = d < 0 ? -d : d;
        if (r + d >= 0 && r + d < IMG_H) Sr  += ex[ad];
        if (c + d >= 0 && c + d < IMG_W) Scl += ex[ad];
    }
    normp[idx] = rsqrtf(Sr * Scl + EPSV);
}

// ---------------------------------------------------------------------------
// Kernel 3: log_softmax over the 21 classes (pred0).
// ---------------------------------------------------------------------------
__global__ void log_softmax_kernel(const float* __restrict__ unary,
                                   float* __restrict__ pred)
{
    int idx = blockIdx.x * blockDim.x + threadIdx.x;     // b*H*W
    if (idx >= NB * IMG_H * IMG_W) return;
    int b = idx / (IMG_H * IMG_W);
    int pix = idx % (IMG_H * IMG_W);
    const float* u = unary + (size_t)b * NCLS * IMG_H * IMG_W + pix;
    float v[NCLS];
    float mx = -INFINITY;
    #pragma unroll
    for (int c = 0; c < NCLS; ++c) { v[c] = u[(size_t)c * IMG_H * IMG_W]; mx = fmaxf(mx, v[c]); }
    float sum = 0.f;
    #pragma unroll
    for (int c = 0; c < NCLS; ++c) sum += __expf(v[c] - mx);
    float lse = mx + __logf(sum);
    float* o = pred + (size_t)b * NCLS * IMG_H * IMG_W + pix;
    #pragma unroll
    for (int c = 0; c < NCLS; ++c) o[(size_t)c * IMG_H * IMG_W] = v[c] - lse;
}

// ---------------------------------------------------------------------------
// Kernel 4: one mean-field iteration. 16x16 tile + 3-halo in LDS, zero-filled
// halos implement the OOB mask for both gaussians. Each thread keeps 21
// accumulator pairs in registers: per tap -> 1 LDS load + 2 FMA per class.
//   msg = normP*sum(wpos[n]*normP_nb*pred_nb) + normC*sum(gcol[n]*normC_nb*pred_nb)
//   out = 0.2*pred + 0.8*msg
// ---------------------------------------------------------------------------
__global__ __launch_bounds__(256)
void crf_iter(const float* __restrict__ predin,
              const float* __restrict__ gcol,
              const float* __restrict__ normc,
              const float* __restrict__ normp,
              const float* __restrict__ col_sdims,
              const float* __restrict__ pos_compat,
              float* __restrict__ predout)
{
    __shared__ float sPred[NCLS][THALO][THALO + 1];
    __shared__ float sNP[THALO][THALO + 1];
    __shared__ float sNC[THALO][THALO + 1];

    const int tx = threadIdx.x, ty = threadIdx.y;
    const int tid = ty * TS + tx;
    const int b  = blockIdx.z;
    const int r0 = blockIdx.y * TS, c0 = blockIdx.x * TS;

    for (int i = tid; i < THALO * THALO; i += 256) {
        int lr = i / THALO, lc = i % THALO;
        int rr = r0 + lr - SPANR, cc = c0 + lc - SPANR;
        bool ok = (rr >= 0 && rr < IMG_H && cc >= 0 && cc < IMG_W);
        sNP[lr][lc] = ok ? normp[rr * IMG_W + cc] : 0.f;
        sNC[lr][lc] = ok ? normc[((size_t)b * IMG_H + rr) * IMG_W + cc] : 0.f;
    }
    for (int i = tid; i < NCLS * THALO * THALO; i += 256) {
        int cl = i / (THALO * THALO);
        int j  = i % (THALO * THALO);
        int lr = j / THALO, lc = j % THALO;
        int rr = r0 + lr - SPANR, cc = c0 + lc - SPANR;
        bool ok = (rr >= 0 && rr < IMG_H && cc >= 0 && cc < IMG_W);
        sPred[cl][lr][lc] = ok ? predin[(((size_t)b * NCLS + cl) * IMG_H + rr) * IMG_W + cc] : 0.f;
    }
    __syncthreads();

    const int r = r0 + ty, c = c0 + tx;
    float s = col_sdims[0];
    float a = 0.5f * s * s;
    float ex[4] = {1.f, __expf(-a), __expf(-4.f*a), __expf(-9.f*a)};
    float cp = pos_compat[0];

    float accP[NCLS], accC[NCLS];
    #pragma unroll
    for (int k = 0; k < NCLS; ++k) { accP[k] = 0.f; accC[k] = 0.f; }

    const float* gp = gcol + (size_t)b * NTAP * IMG_H * IMG_W + (size_t)r * IMG_W + c;

    #pragma unroll 1
    for (int dr = -SPANR; dr <= SPANR; ++dr) {
        int adr = dr < 0 ? -dr : dr;
        // prefetch next row of gaussian planes into cache (CDNA5 global_prefetch_b8)
        if (dr < SPANR)
            __builtin_prefetch(gp + (size_t)((dr + SPANR + 1) * KS) * IMG_H * IMG_W, 0, 1);
        #pragma unroll
        for (int dc = -SPANR; dc <= SPANR; ++dc) {
            int adc = dc < 0 ? -dc : dc;
            int lr = ty + SPANR + dr, lc = tx + SPANR + dc;
            int n = (dr + SPANR) * KS + (dc + SPANR);
            float g  = gp[(size_t)n * IMG_H * IMG_W];
            float w1 = cp * ex[adr] * ex[adc] * sNP[lr][lc];
            float w2 = g * sNC[lr][lc];
            #pragma unroll
            for (int k = 0; k < NCLS; ++k) {
                float t = sPred[k][lr][lc];
                accP[k] = fmaf(w1, t, accP[k]);
                accC[k] = fmaf(w2, t, accC[k]);
            }
        }
    }

    float np0 = sNP[ty + SPANR][tx + SPANR];
    float nc0 = sNC[ty + SPANR][tx + SPANR];
    float* o = predout + ((size_t)b * NCLS * IMG_H + r) * IMG_W + c;
    #pragma unroll
    for (int k = 0; k < NCLS; ++k) {
        float oldv = sPred[k][ty + SPANR][tx + SPANR];
        float msg  = np0 * accP[k] + nc0 * accC[k];
        o[(size_t)k * IMG_H * IMG_W] = (1.f - MOM) * oldv + MOM * msg;
    }
}

// ---------------------------------------------------------------------------
extern "C" void kernel_launch(void* const* d_in, const int* in_sizes, int n_in,
                              void* d_out, int out_size, void* d_ws, size_t ws_size,
                              hipStream_t stream) {
    const float* img        = (const float*)d_in[0];
    const float* unary      = (const float*)d_in[1];
    /* d_in[2] = mesh: exact meshgrid -> pos gaussian handled analytically */
    const float* col_sdims  = (const float*)d_in[4];
    const float* pos_compat = (const float*)d_in[5];
    const float* col_compat = (const float*)d_in[6];
    const float* col_schan  = (const float*)d_in[7];

    float* ws    = (float*)d_ws;
    float* gcol  = ws;                                              // 4*49*H*W
    float* normc = gcol  + (size_t)NB * NTAP * IMG_H * IMG_W;       // 4*H*W
    float* normp = normc + (size_t)NB * IMG_H * IMG_W;              // H*W
    float* predA = normp + (size_t)IMG_H * IMG_W;                   // 4*21*H*W
    float* predB = predA + (size_t)NB * NCLS * IMG_H * IMG_W;       // 4*21*H*W
    float* outp  = (float*)d_out;

    dim3 g1(IMG_W / 16, IMG_H, NB);
    build_col_gaussian<<<g1, 32, 0, stream>>>(img, col_schan, col_compat, gcol, normc);

    build_pos_norm<<<(IMG_H * IMG_W + 255) / 256, 256, 0, stream>>>(col_sdims, normp);

    log_softmax_kernel<<<(NB * IMG_H * IMG_W + 255) / 256, 256, 0, stream>>>(unary, predA);

    dim3 g3(IMG_W / TS, IMG_H / TS, NB);
    dim3 b3(TS, TS);
    crf_iter<<<g3, b3, 0, stream>>>(predA, gcol, normc, normp, col_sdims, pos_compat, predB);
    crf_iter<<<g3, b3, 0, stream>>>(predB, gcol, normc, normp, col_sdims, pos_compat, predA);
    crf_iter<<<g3, b3, 0, stream>>>(predA, gcol, normc, normp, col_sdims, pos_compat, predB);
    crf_iter<<<g3, b3, 0, stream>>>(predB, gcol, normc, normp, col_sdims, pos_compat, predA);
    crf_iter<<<g3, b3, 0, stream>>>(predA, gcol, normc, normp, col_sdims, pos_compat, outp);
}
